// NonLocal3d_37314675868168
// MI455X (gfx1250) — matrix operand order repaired and checked by reading, not compiled, not run on hardware
//
#include <hip/hip_runtime.h>
#include <hip/hip_bf16.h>

typedef _Float16 f16;
typedef __attribute__((ext_vector_type(16))) _Float16 v16h;
typedef __attribute__((ext_vector_type(8)))  _Float16 v8h;
typedef __attribute__((ext_vector_type(8)))  float    v8f;
typedef __attribute__((ext_vector_type(4))) unsigned int u32x4;
typedef __attribute__((ext_vector_type(8))) int          i32x8;
typedef __attribute__((ext_vector_type(4))) int          i32x4;

#ifndef USE_TDM
#define USE_TDM 1
#endif

namespace {
constexpr int B_ = 2, C_ = 256, CI = 128, T_ = 8, H_ = 32, W_ = 32;
constexpr int N_ = T_ * H_ * W_;             // 8192 queries
constexpr int M_ = T_ * (H_/2) * (W_/2);     // 2048 keys (after 2x2 pool)
}

// ---- WMMA fragment helpers (16x16x32 f16, wave32) -------------------------
__device__ __forceinline__ v16h wmma_a_frag(const f16* src, int ld, int lane) {
  const int m = lane & 15, hf = lane >> 4;
  const f16* p = src + m * ld + 8 * hf;
  v8h lo = *(const v8h*)(p);
  v8h hi = *(const v8h*)(p + 16);
  return __builtin_shufflevector(lo, hi, 0,1,2,3,4,5,6,7,8,9,10,11,12,13,14,15);
}
__device__ __forceinline__ v16h wmma_b_frag(const f16* src, int ld, int lane) {
  const int n = lane & 15, hf = lane >> 4;
  v16h r;
  #pragma unroll
  for (int s = 0; s < 8; ++s) {
    r[s]     = src[(8 * hf + s) * ld + n];
    r[s + 8] = src[(16 + 8 * hf + s) * ld + n];
  }
  return r;
}
__device__ __forceinline__ v8f wmma32(v16h a, v16h b, v8f c) {
  return __builtin_amdgcn_wmma_f32_16x16x32_f16(false, a, false, b, (short)0, c,
                                                false, false);
}

// ---- prep: f32 -> f16 conversions ----------------------------------------
__global__ void k_cvt_x(const float* __restrict__ x, f16* __restrict__ xh) {
  size_t i = (size_t)blockIdx.x * 256 + threadIdx.x;
  if (i < (size_t)B_ * C_ * N_) xh[i] = (f16)x[i];
}

__global__ void k_prep_w(const float* __restrict__ theta_w,
                         const float* __restrict__ phi_w,
                         const float* __restrict__ g_w,
                         const float* __restrict__ out_w,
                         f16* __restrict__ wq, f16* __restrict__ wk,
                         f16* __restrict__ wv, f16* __restrict__ owT,
                         float* __restrict__ stats) {
  int i = blockIdx.x * 256 + threadIdx.x;
  const int WSZ = CI * C_;
  if (i < WSZ) {
    wq[i] = (f16)theta_w[i];
    wk[i] = (f16)phi_w[i];
    wv[i] = (f16)g_w[i];
    const int co = i / CI, ci = i % CI;          // out_w is [C, Ci]
    owT[ci * C_ + co] = (f16)out_w[i];           // -> [Ci, C]
  }
  if (i < 2 * C_) stats[i] = 0.f;
}

// ---- fused theta/phi/g 1x1x1 convs: out[Ci,N] = W[Ci,C] x X[C,N] ----------
// Block = (b, which, nt); the 8 waves cover the 8 Ci-tiles and SHARE the
// [C=256 x 16] x-strip, staged once into LDS by the Tensor Data Mover.
__global__ __launch_bounds__(256) void k_qkv(
    const f16* __restrict__ xh,
    const f16* __restrict__ wq, const f16* __restrict__ wk,
    const f16* __restrict__ wv,
    const float* __restrict__ bq, const float* __restrict__ bk,
    const float* __restrict__ bv,
    f16* __restrict__ theta_h,   // [B, N, Ci]  (transposed store)
    f16* __restrict__ phi_f,     // [B, Ci, N]
    f16* __restrict__ g_f) {     // [B, Ci, N]
  __shared__ f16 s_x[C_ * 16];   // 8 KB; ONLY shared object -> LDS offset 0
  const int lane = threadIdx.x & 31, wave = threadIdx.x >> 5;
  int g = blockIdx.x;
  const int nt    = g % (N_ / 16);  g /= (N_ / 16);
  const int which = g % 3;
  const int b     = g / 3;
  const int cit   = wave;          // wave -> Ci tile

  const f16* xg = xh + (size_t)b * C_ * N_ + nt * 16;   // tile start, ld = N

#if USE_TDM && __has_builtin(__builtin_amdgcn_tensor_load_to_lds)
  if (wave == 0) {
    // Tensor DMA descriptor (ISA 8.3/8.4): 2D tile 16 x 256, elem 2B,
    // row stride N elements, dest LDS offset 0.
    const unsigned long long ga = (unsigned long long)(uintptr_t)xg;
    u32x4 g0 = { 1u,                                   // count=1, user mode
                 0u,                                   // lds_addr = 0
                 (unsigned)(ga & 0xFFFFFFFFu),         // global_addr lo
                 (unsigned)((ga >> 32) & 0x01FFFFFFu)  // global_addr hi
                   | (2u << 30) };                     // type=2 (image)
    i32x8 g1 = { (int)(1u << 16),                          // data_size=2B
                 (int)(((unsigned)N_ & 0xFFFFu) << 16),    // tensor_dim0 lo16
                 (int)(((unsigned)N_ >> 16)                // tensor_dim0 hi16
                       | (((unsigned)C_ & 0xFFFFu) << 16)),//  | tensor_dim1 lo16
                 (int)(((unsigned)C_ >> 16) | (16u << 16)),// td1 hi | tile_dim0=16
                 (int)(unsigned)C_,                        // tile_dim1=C, tile_dim2=0
                 (int)(unsigned)N_,                        // dim0 stride lo32
                 0, 0 };                                   // stride hi, dim1 stride
    i32x4 zz4 = { 0, 0, 0, 0 };                            // groups 2/3 unused (2D)
    i32x8 zz8 = { 0, 0, 0, 0, 0, 0, 0, 0 };
    __builtin_amdgcn_tensor_load_to_lds(g0, g1, zz4, zz4, zz8, 0);
  }
  // The TDM writes LDS behind the compiler's back; make sure the compiler
  // cannot prove s_x is unwritten (round-4/5 asm folded the B-fragment
  // ds_loads to undef):
  //  (a) never-taken guard store plants a real store to s_x in the module
  //      (grid is 3072 blocks, so wgid never hits 0x7FFFFFFF; uniform branch
  //       keeps EXEC all-ones for the WMMAs below);
  //  (b) escape s_x into the wait-asm so the memory clobber must alias it.
  if (__builtin_amdgcn_workgroup_id_x() == 0x7FFFFFFFu)
    s_x[threadIdx.x] = (f16)0.f;
  asm volatile("s_wait_tensorcnt 0x0" :: "v"((const f16*)s_x) : "memory");
#else
  for (int r = threadIdx.x; r < C_; r += 256)
    *(v16h*)&s_x[r * 16] = *(const v16h*)(xg + (size_t)r * N_);
#endif
  __syncthreads();

  const f16*   wsel = which == 0 ? wq : (which == 1 ? wk : wv);
  const float* bsel = which == 0 ? bq : (which == 1 ? bk : bv);
  const f16* A0 = wsel + cit * 16 * C_;                       // ld = C

  // dual accumulators: break the WMMA->WMMA D->C dependency chain
  v8f acc0 = {}, acc1 = {};
  #pragma unroll
  for (int k = 0; k < C_; k += 64) {
    acc0 = wmma32(wmma_a_frag(A0 + k, C_, lane),
                  wmma_b_frag(s_x + k * 16, 16, lane), acc0);
    acc1 = wmma32(wmma_a_frag(A0 + k + 32, C_, lane),
                  wmma_b_frag(s_x + (k + 32) * 16, 16, lane), acc1);
  }
  v8f acc = acc0 + acc1;

  const int hf = lane >> 4, col = lane & 15;
  #pragma unroll
  for (int j = 0; j < 8; ++j) acc[j] += bsel[cit * 16 + j + 8 * hf];

  if (which == 0) {  // transposed: theta_h[n, ci]
    f16* dst = theta_h + (size_t)b * N_ * CI + (size_t)(nt * 16) * CI + cit * 16;
    #pragma unroll
    for (int j = 0; j < 8; ++j) dst[(size_t)col * CI + (j + 8 * hf)] = (f16)acc[j];
  } else {           // natural: [ci, n]
    f16* dst = (which == 1 ? phi_f : g_f) +
               (size_t)b * CI * N_ + (size_t)(cit * 16) * N_ + nt * 16;
    #pragma unroll
    for (int j = 0; j < 8; ++j) dst[(size_t)(j + 8 * hf) * N_ + col] = (f16)acc[j];
  }
}

// ---- MaxPool3d (1,2,2) for phi & g; also lay out for attention B-operands -
__global__ void k_pool(const f16* __restrict__ phi_f, const f16* __restrict__ g_f,
                       f16* __restrict__ phi_h,   // [B, Ci, M]
                       f16* __restrict__ g_ht) {  // [B, M, Ci]
  int i = blockIdx.x * 256 + threadIdx.x;
  if (i >= B_ * CI * M_) return;
  const int m  = i % M_;
  const int ci = (i / M_) % CI;
  const int b  = i / (M_ * CI);
  const int t = m >> 8, hp = (m >> 4) & 15, wp = m & 15;
  const int n0 = t * (H_ * W_) + hp * 2 * W_ + wp * 2;
  const size_t base = (size_t)b * CI * N_ + (size_t)ci * N_;
  const f16* pf = phi_f + base;
  const f16* gf = g_f + base;
  float p = fmaxf(fmaxf((float)pf[n0], (float)pf[n0 + 1]),
                  fmaxf((float)pf[n0 + W_], (float)pf[n0 + W_ + 1]));
  float g = fmaxf(fmaxf((float)gf[n0], (float)gf[n0 + 1]),
                  fmaxf((float)gf[n0 + W_], (float)gf[n0 + W_ + 1]));
  phi_h[(size_t)b * CI * M_ + (size_t)ci * M_ + m] = (f16)p;
  g_ht[(size_t)b * M_ * CI + (size_t)m * CI + ci] = (f16)g;
}

// ---- attention: per block, 16 query rows; full 16x2048 score strip in LDS -
__global__ __launch_bounds__(256) void k_attn(
    const f16* __restrict__ theta_h,  // [B, N, Ci]
    const f16* __restrict__ phi_h,    // [B, Ci, M]
    const f16* __restrict__ g_ht,     // [B, M, Ci]
    f16* __restrict__ y_h) {          // [B, N, Ci]
  __shared__ float s_s[16 * M_];      // 128 KB scores (f32)
  __shared__ f16   s_p[16 * M_];      //  64 KB normalized P (f16)
  const int lane = threadIdx.x & 31, wave = threadIdx.x >> 5;
  const int b  = blockIdx.x / (N_ / 16);
  const int n0 = (blockIdx.x % (N_ / 16)) * 16;
  const int hf = lane >> 4, colL = lane & 15;

  // preload the 16x128 theta A-operand (shared by all score tiles)
  const f16* th = theta_h + (size_t)b * N_ * CI + (size_t)n0 * CI;
  v16h aq[4];
  #pragma unroll
  for (int kk = 0; kk < 4; ++kk) aq[kk] = wmma_a_frag(th + kk * 32, CI, lane);

  // phase 1: S = theta x phi   (each wave: 16 of the 128 m-tiles)
  const f16* ph = phi_h + (size_t)b * CI * M_;
  for (int mt = wave; mt < M_ / 16; mt += 8) {
    v8f acc0 = {}, acc1 = {};
    acc0 = wmma32(aq[0], wmma_b_frag(ph + (size_t)(0 * 32) * M_ + mt * 16, M_, lane), acc0);
    acc1 = wmma32(aq[1], wmma_b_frag(ph + (size_t)(1 * 32) * M_ + mt * 16, M_, lane), acc1);
    acc0 = wmma32(aq[2], wmma_b_frag(ph + (size_t)(2 * 32) * M_ + mt * 16, M_, lane), acc0);
    acc1 = wmma32(aq[3], wmma_b_frag(ph + (size_t)(3 * 32) * M_ + mt * 16, M_, lane), acc1);
    v8f acc = acc0 + acc1;
    const int col = mt * 16 + colL;
    #pragma unroll
    for (int j = 0; j < 8; ++j) s_s[(j + 8 * hf) * M_ + col] = acc[j];
  }
  __syncthreads();

  // phase 2: row softmax in LDS (each wave owns 2 rows)
  for (int r = wave * 2; r < wave * 2 + 2; ++r) {
    float mx = -3.4e38f;
    for (int k = lane; k < M_; k += 32) mx = fmaxf(mx, s_s[r * M_ + k]);
    #pragma unroll
    for (int off = 16; off > 0; off >>= 1) mx = fmaxf(mx, __shfl_xor(mx, off, 32));
    float sum = 0.f;
    for (int k = lane; k < M_; k += 32) {
      float e = __expf(s_s[r * M_ + k] - mx);
      s_s[r * M_ + k] = e;
      sum += e;
    }
    #pragma unroll
    for (int off = 16; off > 0; off >>= 1) sum += __shfl_xor(sum, off, 32);
    const float inv = __frcp_rn(sum);
    for (int k = lane; k < M_; k += 32)
      s_p[r * M_ + k] = (f16)(s_s[r * M_ + k] * inv);
  }
  __syncthreads();

  // phase 3: y = P x g   (each wave: one 16x16 tile of the 16x128 result)
  v8f acc0 = {}, acc1 = {};
  const f16* gv = g_ht + (size_t)b * M_ * CI + wave * 16;
  for (int kt = 0; kt < M_; kt += 64) {
    acc0 = wmma32(wmma_a_frag(s_p + kt, M_, lane),
                  wmma_b_frag(gv + (size_t)kt * CI, CI, lane), acc0);
    acc1 = wmma32(wmma_a_frag(s_p + kt + 32, M_, lane),
                  wmma_b_frag(gv + (size_t)(kt + 32) * CI, CI, lane), acc1);
  }
  v8f acc = acc0 + acc1;
  f16* dst = y_h + (size_t)b * N_ * CI + (size_t)n0 * CI + wave * 16;
  #pragma unroll
  for (int j = 0; j < 8; ++j) dst[(size_t)(j + 8 * hf) * CI + colL] = (f16)acc[j];
}

// ---- out conv + BN batch statistics ---------------------------------------
__global__ __launch_bounds__(256) void k_outconv(
    const f16* __restrict__ y_h,   // [B, N, Ci]
    const f16* __restrict__ owT,   // [Ci, C]
    const float* __restrict__ out_b,
    float* __restrict__ z,         // [B, N, C]
    float* __restrict__ stats) {   // [2*C]: sum, sumsq
  const int lane = threadIdx.x & 31, wave = threadIdx.x >> 5;
  int g = blockIdx.x * 8 + wave;
  const int ct = g % (C_ / 16);  g /= (C_ / 16);
  const int nt = g % (N_ / 16);
  const int b  = g / (N_ / 16);

  const f16* A0 = y_h + (size_t)b * N_ * CI + (size_t)(nt * 16) * CI;  // ld=Ci
  const f16* B0 = owT + ct * 16;                                       // ld=C
  v8f acc0 = {}, acc1 = {};
  #pragma unroll
  for (int k = 0; k < CI; k += 64) {
    acc0 = wmma32(wmma_a_frag(A0 + k, CI, lane),
                  wmma_b_frag(B0 + (size_t)k * C_, C_, lane), acc0);
    acc1 = wmma32(wmma_a_frag(A0 + k + 32, CI, lane),
                  wmma_b_frag(B0 + (size_t)(k + 32) * C_, C_, lane), acc1);
  }
  v8f acc = acc0 + acc1;

  const int hf = lane >> 4, col = lane & 15;
  const int co = ct * 16 + col;
  const float bias = out_b[co];
  float s = 0.f, q = 0.f;
  float* dst = z + (size_t)b * N_ * C_ + (size_t)(nt * 16) * C_ + ct * 16;
  #pragma unroll
  for (int j = 0; j < 8; ++j) {
    float v = acc[j] + bias;
    dst[(size_t)(j + 8 * hf) * C_ + col] = v;
    s += v;
    q += v * v;
  }
  s += __shfl_xor(s, 16, 32);   // lanes l / l+16 share the same column
  q += __shfl_xor(q, 16, 32);
  if (lane < 16) {
    atomicAdd(&stats[co], s);
    atomicAdd(&stats[C_ + co], q);
  }
}

// ---- BN normalize + affine + residual -------------------------------------
__global__ void k_bn_res(const float* __restrict__ x, const float* __restrict__ z,
                         const float* __restrict__ stats,
                         const float* __restrict__ bn_w,
                         const float* __restrict__ bn_b,
                         float* __restrict__ out) {
  size_t i = (size_t)blockIdx.x * 256 + threadIdx.x;
  if (i >= (size_t)B_ * C_ * N_) return;
  const int n = (int)(i % N_);
  const int c = (int)((i / N_) % C_);
  const int b = (int)(i / ((size_t)N_ * C_));
  const float cnt  = 1.f / (float)(B_ * N_);
  const float mean = stats[c] * cnt;
  const float var  = stats[C_ + c] * cnt - mean * mean;
  const float zz   = z[(size_t)b * N_ * C_ + (size_t)n * C_ + c];
  out[i] = x[i] + (zz - mean) * rsqrtf(var + 1e-5f) * bn_w[c] + bn_b[c];
}

// ---- host-side launch ------------------------------------------------------
extern "C" void kernel_launch(void* const* d_in, const int* in_sizes, int n_in,
                              void* d_out, int out_size, void* d_ws, size_t ws_size,
                              hipStream_t stream) {
  const float* x     = (const float*)d_in[0];
  const float* g_w   = (const float*)d_in[1];
  const float* g_b   = (const float*)d_in[2];
  const float* th_w  = (const float*)d_in[3];
  const float* th_b  = (const float*)d_in[4];
  const float* ph_w  = (const float*)d_in[5];
  const float* ph_b  = (const float*)d_in[6];
  const float* out_w = (const float*)d_in[7];
  const float* out_b = (const float*)d_in[8];
  const float* bn_w  = (const float*)d_in[9];
  const float* bn_b  = (const float*)d_in[10];
  float* out = (float*)d_out;

  char* ws = (char*)d_ws;
  size_t off = 0;
  auto take = [&](size_t bytes) {
    char* p = ws + off;
    off = (off + bytes + 255) & ~(size_t)255;
    return p;
  };
  f16* xh    = (f16*)take((size_t)B_ * C_ * N_ * 2);   // 8.4 MB
  f16* wq    = (f16*)take((size_t)CI * C_ * 2);
  f16* wk    = (f16*)take((size_t)CI * C_ * 2);
  f16* wv    = (f16*)take((size_t)CI * C_ * 2);
  f16* owT   = (f16*)take((size_t)CI * C_ * 2);
  f16* thh   = (f16*)take((size_t)B_ * N_ * CI * 2);   // 4.2 MB
  f16* phf   = (f16*)take((size_t)B_ * CI * N_ * 2);   // 4.2 MB
  f16* gf    = (f16*)take((size_t)B_ * CI * N_ * 2);   // 4.2 MB
  f16* phh   = (f16*)take((size_t)B_ * CI * M_ * 2);   // 1.0 MB
  f16* ght   = (f16*)take((size_t)B_ * M_ * CI * 2);   // 1.0 MB
  f16* yh    = (f16*)take((size_t)B_ * N_ * CI * 2);   // 4.2 MB
  float* z   = (float*)take((size_t)B_ * N_ * C_ * 4); // 16.8 MB
  float* st  = (float*)take((size_t)2 * C_ * 4);
  (void)ws_size;

  const int elems = B_ * C_ * N_;                      // 4.19 M
  k_cvt_x<<<(elems + 255) / 256, 256, 0, stream>>>(x, xh);
  k_prep_w<<<(CI * C_ + 255) / 256, 256, 0, stream>>>(th_w, ph_w, g_w, out_w,
                                                      wq, wk, wv, owT, st);

  // block = (b, which, nt); 8 waves cover the 8 Ci tiles
  k_qkv<<<B_ * 3 * (N_ / 16), 256, 0, stream>>>(xh, wq, wk, wv, th_b, ph_b, g_b,
                                                thh, phf, gf);

  k_pool<<<(B_ * CI * M_ + 255) / 256, 256, 0, stream>>>(phf, gf, phh, ght);

  k_attn<<<B_ * (N_ / 16), 256, 0, stream>>>(thh, phh, ght, yh);

  const int oc_tiles = B_ * (N_ / 16) * (C_ / 16);           // 16384
  k_outconv<<<oc_tiles / 8, 256, 0, stream>>>(yh, owT, out_b, z, st);

  k_bn_res<<<(elems + 255) / 256, 256, 0, stream>>>(x, z, st, bn_w, bn_b, out);
  (void)in_sizes; (void)n_in; (void)out_size;
}
